// MultiHeadSelfAttention_24111946399776
// MI455X (gfx1250) — matrix operand order repaired and checked by reading, not compiled, not run on hardware
//
#include <hip/hip_runtime.h>

// ---------------------------------------------------------------------------
// MHSA for MI455X (gfx1250): WMMA bf16 matmuls + async-to-LDS K/V staging
// ---------------------------------------------------------------------------

typedef __bf16 bf16_t;
typedef __attribute__((ext_vector_type(8)))  __bf16 v8bf;
typedef __attribute__((ext_vector_type(16))) __bf16 v16bf;
typedef __attribute__((ext_vector_type(8)))  float  v8f;

#define CDIM 1024
#define NDIM 2048
#define BDIM 4
#define HDIM 16
#define DHDIM 64

__device__ __forceinline__ bf16_t f2bf(float x) {
  union { float f; unsigned u; } v; v.f = x;
  unsigned r = v.u + 0x7FFFu + ((v.u >> 16) & 1u);   // round-to-nearest-even
  unsigned short h = (unsigned short)(r >> 16);
  union { unsigned short s; bf16_t b; } o; o.s = h;
  return o.b;
}

// A-fragment (16x32 bf16, MxK): lane r=lane&15 supplies row r.
// lanes 0-15 hold K = {0..7, 16..23}; lanes 16-31 hold K = {8..15, 24..31}.
__device__ __forceinline__ v16bf load_frag_a(const bf16_t* base, int ld) {
  int lane = threadIdx.x & 31;
  int r = lane & 15;
  int half = lane >> 4;
  const bf16_t* p = base + (size_t)r * ld + 8 * half;
  v8bf lo = *(const v8bf*)(p);        // K = 8*half .. 8*half+7
  v8bf hi = *(const v8bf*)(p + 16);   // K = 16+8*half .. 16+8*half+7
  return __builtin_shufflevector(lo, hi, 0,1,2,3,4,5,6,7,8,9,10,11,12,13,14,15);
}

// B-fragment (32x16 bf16, KxN): lane n=lane&15 supplies column n.
// lanes 0-15 hold K = {0..7, 16..23}; lanes 16-31 hold K = {8..15, 24..31}.
__device__ __forceinline__ v16bf load_frag_b(const bf16_t* base, int ld) {
  int lane = threadIdx.x & 31;
  int n = lane & 15;
  int half = lane >> 4;
  const bf16_t* p = base + (size_t)n * ld + 8 * half;
  v8bf lo = *(const v8bf*)(p);
  v8bf hi = *(const v8bf*)(p + 16);
  return __builtin_shufflevector(lo, hi, 0,1,2,3,4,5,6,7,8,9,10,11,12,13,14,15);
}

__device__ __forceinline__ v8f wmma_bf16(v16bf a, v16bf b, v8f c) {
  return __builtin_amdgcn_wmma_f32_16x16x32_bf16(false, a, false, b,
                                                 (short)0, c, false, false);
}

// gfx1250 async DMA: global -> LDS, 16 bytes per lane, tracked by ASYNCcnt.
__device__ __forceinline__ void async_load_b128(unsigned lds_byte_off,
                                                const void* gaddr) {
  asm volatile("global_load_async_to_lds_b128 %0, %1, off"
               :: "v"(lds_byte_off), "v"(gaddr) : "memory");
}

__device__ __forceinline__ void wait_async0() {
#if __has_builtin(__builtin_amdgcn_s_wait_asynccnt)
  __builtin_amdgcn_s_wait_asynccnt(0);
#else
  asm volatile("s_wait_asynccnt 0" ::: "memory");
#endif
}

// ---------------------------------------------------------------------------
// Kernel 1: LayerNorm (f32 in -> bf16 out)
// ---------------------------------------------------------------------------
__global__ __launch_bounds__(256)
void mhsa_ln_kernel(const float* __restrict__ x, const float* __restrict__ g,
                    const float* __restrict__ beta, bf16_t* __restrict__ h) {
  int row = blockIdx.x;
  const float* xr = x + (size_t)row * CDIM;
  bf16_t* hr = h + (size_t)row * CDIM;
  float s = 0.f, ss = 0.f;
  for (int i = threadIdx.x; i < CDIM; i += 256) {
    float v = xr[i]; s += v; ss += v * v;
  }
  __shared__ float r0[256], r1[256];
  r0[threadIdx.x] = s; r1[threadIdx.x] = ss;
  __syncthreads();
  for (int off = 128; off > 0; off >>= 1) {
    if (threadIdx.x < off) {
      r0[threadIdx.x] += r0[threadIdx.x + off];
      r1[threadIdx.x] += r1[threadIdx.x + off];
    }
    __syncthreads();
  }
  float mean = r0[0] * (1.0f / CDIM);
  float var  = r1[0] * (1.0f / CDIM) - mean * mean;
  float inv  = rsqrtf(var + 1e-5f);
  for (int i = threadIdx.x; i < CDIM; i += 256)
    hr[i] = f2bf((xr[i] - mean) * inv * g[i] + beta[i]);
}

// ---------------------------------------------------------------------------
// Kernel 2: f32 -> bf16 conversion (weights)
// ---------------------------------------------------------------------------
__global__ __launch_bounds__(256)
void mhsa_cvt_kernel(const float* __restrict__ in, bf16_t* __restrict__ out, int n) {
  int i = blockIdx.x * 256 + threadIdx.x;
  if (i < n) out[i] = f2bf(in[i]);
}

// ---------------------------------------------------------------------------
// Kernel 3: QKV GEMM  qkv = h @ Wqkv^T  -> scatter into q (scaled), k, v^T
// Wave computes 32x64 (2 A-frags x 4 B-frags, 8 accumulators).
// grid (8192/32, 3072/512), 256 threads (8 waves)
// ---------------------------------------------------------------------------
__global__ __launch_bounds__(256)
void mhsa_qkv_gemm_kernel(const bf16_t* __restrict__ h,  // [8192,1024]
                          const bf16_t* __restrict__ w,  // [3072,1024]
                          bf16_t* __restrict__ qh,       // [B*H, N, Dh] (pre-scaled)
                          bf16_t* __restrict__ kh,       // [B*H, N, Dh]
                          bf16_t* __restrict__ vt) {     // [B*H, Dh, N]
  int wave = threadIdx.x >> 5;
  int lane = threadIdx.x & 31;
  int m0 = blockIdx.x * 32;
  int n0 = blockIdx.y * 512 + wave * 64;
  const bf16_t* A0 = h + (size_t)m0 * CDIM;
  const bf16_t* A1 = A0 + (size_t)16 * CDIM;
  const bf16_t* B0 = w + (size_t)n0 * CDIM;
  v8f acc[2][4];
#pragma unroll
  for (int i = 0; i < 2; ++i)
#pragma unroll
    for (int j = 0; j < 4; ++j) acc[i][j] = {};

  for (int k = 0; k < CDIM; k += 32) {
    v16bf a0 = load_frag_a(A0 + k, CDIM);
    v16bf a1 = load_frag_a(A1 + k, CDIM);
#pragma unroll
    for (int j = 0; j < 4; ++j) {
      v16bf b = load_frag_b(B0 + (size_t)(16 * j) * CDIM + k, CDIM);
      acc[0][j] = wmma_bf16(a0, b, acc[0][j]);
      acc[1][j] = wmma_bf16(a1, b, acc[1][j]);
    }
  }

  int half = lane >> 4, col = lane & 15;
#pragma unroll
  for (int i = 0; i < 2; ++i) {
#pragma unroll
    for (int j = 0; j < 4; ++j) {
#pragma unroll
      for (int e = 0; e < 8; ++e) {
        int m = m0 + 16 * i + e + 8 * half;      // global row in [0, B*N)
        int d = n0 + 16 * j + col;               // output dim in [0, 3C)
        int bidx = m >> 11, npos = m & (NDIM - 1);
        int sel = d >> 10;                       // 0=q 1=k 2=v
        int c = d & (CDIM - 1);
        int head = c >> 6, dh = c & (DHDIM - 1);
        size_t bhh = (size_t)bidx * HDIM + head;
        float v = acc[i][j][e];
        if (sel == 0)      qh[(bhh * NDIM + npos) * DHDIM + dh] = f2bf(v * 0.125f);
        else if (sel == 1) kh[(bhh * NDIM + npos) * DHDIM + dh] = f2bf(v);
        else               vt[(bhh * DHDIM + dh) * NDIM + npos] = f2bf(v);
      }
    }
  }
}

// ---------------------------------------------------------------------------
// Kernel 4: flash attention per (b,h). Block = 8 waves x 16 query rows.
// K/V tiles (32 keys) double-buffered in LDS via global_load_async_to_lds_b128,
// shared by all waves; P re-packed through LDS into A-fragment layout.
// grid (N/128, B*H), 256 threads
// ---------------------------------------------------------------------------
__global__ __launch_bounds__(256)
void mhsa_attn_kernel(const bf16_t* __restrict__ qh, const bf16_t* __restrict__ kh,
                      const bf16_t* __restrict__ vt, bf16_t* __restrict__ out) {
  __shared__ __align__(32) bf16_t kbuf[2][32 * DHDIM];   // [key][dh]   4KB each
  __shared__ __align__(32) bf16_t vbuf[2][DHDIM * 32];   // [dh][key]   4KB each
  __shared__ __align__(32) bf16_t pbuf[8][16 * 32];      // per-wave P  8KB

  int wave = threadIdx.x >> 5;
  int lane = threadIdx.x & 31;
  int half = lane >> 4, col = lane & 15;
  int bh = blockIdx.y;                       // b*H + h
  int b = bh >> 4, hd = bh & 15;
  int q0 = blockIdx.x * 128 + wave * 16;

  const bf16_t* qbase = qh + ((size_t)bh * NDIM + q0) * DHDIM;
  const bf16_t* kbase = kh + (size_t)bh * NDIM * DHDIM;
  const bf16_t* vbase = vt + (size_t)bh * DHDIM * NDIM;

  // stage one 32-key K tile (contiguous 4KB) + V tile (64 rows x 64B) to LDS
  auto issue_tile = [&](int buf, int kt) {
    unsigned t = threadIdx.x;
    const bf16_t* gk = kbase + (size_t)kt * DHDIM + t * 8;          // 16B per lane
    unsigned lk = (unsigned)(size_t)(&kbuf[buf][0]) + t * 16;
    async_load_b128(lk, gk);
    unsigned row = t >> 2, c4 = t & 3;
    const bf16_t* gv = vbase + (size_t)row * NDIM + kt + c4 * 8;
    unsigned lv = (unsigned)(size_t)(&vbuf[buf][0]) + row * 64 + c4 * 16;
    async_load_b128(lv, gv);
  };

  v16bf aq0 = load_frag_a(qbase + 0,  DHDIM);   // Q[:, 0:32]
  v16bf aq1 = load_frag_a(qbase + 32, DHDIM);   // Q[:, 32:64]

  v8f o0 = {}, o1 = {}, o2 = {}, o3 = {};
  float mrow[8], lrow[8];
#pragma unroll
  for (int j = 0; j < 8; ++j) { mrow[j] = -1e30f; lrow[j] = 0.f; }

  bf16_t* myp = pbuf[wave];
  issue_tile(0, 0);

  for (int kt = 0; kt < NDIM; kt += 32) {
    int cur = (kt >> 5) & 1;
    wait_async0();
    __syncthreads();                       // tile visible to all waves
    if (kt + 32 < NDIM) issue_tile(cur ^ 1, kt + 32);

    const bf16_t* kl = kbuf[cur];
    const bf16_t* vl = vbuf[cur];

    // ---- S = Q K^T for 32 keys (two 16x16 C tiles), frags from LDS ----
    v8f s0 = {}, s1 = {};
    s0 = wmma_bf16(aq0, load_frag_b(kl + 0,  DHDIM), s0);
    s0 = wmma_bf16(aq1, load_frag_b(kl + 32, DHDIM), s0);
    s1 = wmma_bf16(aq0, load_frag_b(kl + 16 * DHDIM + 0,  DHDIM), s1);
    s1 = wmma_bf16(aq1, load_frag_b(kl + 16 * DHDIM + 32, DHDIM), s1);

    // ---- online softmax update (rows live across the 16-lane half) ----
    float p0[8], p1[8], alpha[8];
#pragma unroll
    for (int j = 0; j < 8; ++j) {
      float mt = fmaxf(s0[j], s1[j]);
#pragma unroll
      for (int off = 1; off < 16; off <<= 1)
        mt = fmaxf(mt, __shfl_xor(mt, off, 32));
      float mnew = fmaxf(mrow[j], mt);
      alpha[j] = __expf(mrow[j] - mnew);
      p0[j] = __expf(s0[j] - mnew);
      p1[j] = __expf(s1[j] - mnew);
      float t = p0[j] + p1[j];
#pragma unroll
      for (int off = 1; off < 16; off <<= 1)
        t += __shfl_xor(t, off, 32);
      lrow[j] = lrow[j] * alpha[j] + t;
      mrow[j] = mnew;
    }
#pragma unroll
    for (int j = 0; j < 8; ++j) {
      o0[j] *= alpha[j]; o1[j] *= alpha[j]; o2[j] *= alpha[j]; o3[j] *= alpha[j];
    }

    // ---- stage P (16x32, C-layout -> row-major LDS -> A-fragment) ----
    __syncthreads();
#pragma unroll
    for (int j = 0; j < 8; ++j) {
      int r = j + 8 * half;
      myp[r * 32 + col]      = f2bf(p0[j]);
      myp[r * 32 + 16 + col] = f2bf(p1[j]);
    }
    __syncthreads();
    v16bf ap = load_frag_a(myp, 32);

    // ---- O += P V : B column n = vbuf row (16t+n), contiguous along keys ----
    o0 = wmma_bf16(ap, load_frag_b(vl + 0 * 16 * 32, 32), o0);
    o1 = wmma_bf16(ap, load_frag_b(vl + 1 * 16 * 32, 32), o1);
    o2 = wmma_bf16(ap, load_frag_b(vl + 2 * 16 * 32, 32), o2);
    o3 = wmma_bf16(ap, load_frag_b(vl + 3 * 16 * 32, 32), o3);
  }

  // ---- finalize: divide by row sum, store bf16 to [B*N, C] for proj gemm ----
#pragma unroll
  for (int j = 0; j < 8; ++j) {
    float inv = 1.0f / lrow[j];
    int m = q0 + j + 8 * half;
    size_t base = ((size_t)b * NDIM + m) * CDIM + (size_t)hd * DHDIM + col;
    out[base +  0] = f2bf(o0[j] * inv);
    out[base + 16] = f2bf(o1[j] * inv);
    out[base + 32] = f2bf(o2[j] * inv);
    out[base + 48] = f2bf(o3[j] * inv);
  }
}

// ---------------------------------------------------------------------------
// Kernel 5: proj GEMM + bias + residual (f32 out). Wave computes 32x64.
// grid (8192/32, 1024/512), 256 threads
// ---------------------------------------------------------------------------
__global__ __launch_bounds__(256)
void mhsa_proj_gemm_kernel(const bf16_t* __restrict__ a,   // attn out [8192,1024]
                           const bf16_t* __restrict__ w,   // [1024,1024]
                           const float* __restrict__ bias,
                           const float* __restrict__ x,    // residual
                           float* __restrict__ out) {
  int wave = threadIdx.x >> 5;
  int lane = threadIdx.x & 31;
  int m0 = blockIdx.x * 32;
  int n0 = blockIdx.y * 512 + wave * 64;
  const bf16_t* A0 = a + (size_t)m0 * CDIM;
  const bf16_t* A1 = A0 + (size_t)16 * CDIM;
  const bf16_t* B0 = w + (size_t)n0 * CDIM;
  v8f acc[2][4];
#pragma unroll
  for (int i = 0; i < 2; ++i)
#pragma unroll
    for (int j = 0; j < 4; ++j) acc[i][j] = {};

  for (int k = 0; k < CDIM; k += 32) {
    v16bf a0 = load_frag_a(A0 + k, CDIM);
    v16bf a1 = load_frag_a(A1 + k, CDIM);
#pragma unroll
    for (int j = 0; j < 4; ++j) {
      v16bf bfr = load_frag_b(B0 + (size_t)(16 * j) * CDIM + k, CDIM);
      acc[0][j] = wmma_bf16(a0, bfr, acc[0][j]);
      acc[1][j] = wmma_bf16(a1, bfr, acc[1][j]);
    }
  }

  int half = lane >> 4, col = lane & 15;
#pragma unroll
  for (int i = 0; i < 2; ++i) {
#pragma unroll
    for (int j = 0; j < 4; ++j) {
      int n = n0 + 16 * j + col;
      float bv = bias[n];
#pragma unroll
      for (int e = 0; e < 8; ++e) {
        int m = m0 + 16 * i + e + 8 * half;
        size_t idx = (size_t)m * CDIM + n;
        out[idx] = acc[i][j][e] + bv + x[idx];
      }
    }
  }
}

// ---------------------------------------------------------------------------
extern "C" void kernel_launch(void* const* d_in, const int* in_sizes, int n_in,
                              void* d_out, int out_size, void* d_ws, size_t ws_size,
                              hipStream_t stream) {
  const float* x       = (const float*)d_in[0];
  const float* ln_g    = (const float*)d_in[1];
  const float* ln_b    = (const float*)d_in[2];
  const float* w_qkv   = (const float*)d_in[3];
  const float* w_proj  = (const float*)d_in[4];
  const float* b_proj  = (const float*)d_in[5];
  float* out = (float*)d_out;

  const size_t ROWS = (size_t)BDIM * NDIM;          // 8192
  size_t off = 0;
  auto carve = [&](size_t bytes) {
    void* p = (char*)d_ws + off;
    off += (bytes + 255) & ~(size_t)255;
    return p;
  };
  bf16_t* h_bf    = (bf16_t*)carve(ROWS * CDIM * 2);
  bf16_t* wqkv_bf = (bf16_t*)carve((size_t)3 * CDIM * CDIM * 2);
  bf16_t* wprj_bf = (bf16_t*)carve((size_t)CDIM * CDIM * 2);
  bf16_t* q_bf    = (bf16_t*)carve(ROWS * CDIM * 2);
  bf16_t* k_bf    = (bf16_t*)carve(ROWS * CDIM * 2);
  bf16_t* vt_bf   = (bf16_t*)carve(ROWS * CDIM * 2);
  bf16_t* attn_bf = (bf16_t*)carve(ROWS * CDIM * 2);
  (void)ws_size; (void)in_sizes; (void)n_in; (void)out_size;

  mhsa_ln_kernel<<<dim3((unsigned)ROWS), 256, 0, stream>>>(x, ln_g, ln_b, h_bf);
  {
    int n1 = 3 * CDIM * CDIM;
    mhsa_cvt_kernel<<<(n1 + 255) / 256, 256, 0, stream>>>(w_qkv, wqkv_bf, n1);
    int n2 = CDIM * CDIM;
    mhsa_cvt_kernel<<<(n2 + 255) / 256, 256, 0, stream>>>(w_proj, wprj_bf, n2);
  }
  mhsa_qkv_gemm_kernel<<<dim3((unsigned)(ROWS / 32), (3 * CDIM) / 512), 256, 0, stream>>>(
      h_bf, wqkv_bf, q_bf, k_bf, vt_bf);
  mhsa_attn_kernel<<<dim3(NDIM / 128, BDIM * HDIM), 256, 0, stream>>>(
      q_bf, k_bf, vt_bf, attn_bf);
  mhsa_proj_gemm_kernel<<<dim3((unsigned)(ROWS / 32), CDIM / 512), 256, 0, stream>>>(
      attn_bf, wprj_bf, b_proj, x, out);
}
